// MAMBA_BayesMAGAC_9474697855322
// MI455X (gfx1250) — matrix operand order repaired and checked by reading, not compiled, not run on hardware
//
#include <hip/hip_runtime.h>

// ---------------------------------------------------------------------------
// MI455X / gfx1250 implementation. wave32, WMMA f32_16x16x32_f16 for all GEMMs.
// Dominant cost: MAGAC Chebyshev einsum = 128 GEMMs of 512x1024x512 (~69 GFLOP)
// fused with the Wf contraction (atomic f32 reduction). Each wave computes a
// 32x32 C tile (4 WMMAs/K-step, fragment reuse). K-loop is double-buffered:
// the Tensor Data Mover (tensor_load_to_lds + s_wait_tensorcnt) fills the next
// A tile while WMMAs consume the current one; B tiles via b128 loads issued
// ahead of the compute into the ping-pong buffer. All GEMM call sites are
// padded to hit the vectorized staging path (x_dbl padded 132 -> 160 cols).
// ---------------------------------------------------------------------------

typedef __attribute__((ext_vector_type(16))) _Float16 v16h;
typedef __attribute__((ext_vector_type(8)))  _Float16 v8h;
typedef __attribute__((ext_vector_type(8)))  float    v8f;

#define BB   8
#define SEQL 1024
#define DM   512
#define DE_  64
#define DH_  64
#define DTR_ 4
#define DU_  32
#define HH   4
#define KCH  3
#define DEMB_ 10
#define NN_  512
#define BL_  (BB * SEQL)      /* 8192 rows */
#define XDS  160              /* padded x_dbl row stride (>=132, mult of 32) */

#define GBM 128
#define GBN 32
#define GBK 32
#define LDP (GBK + 8)         /* LDS row stride in halves; +16B pad per row */

// ----------------------------- TDM (async tensor) ---------------------------
#if defined(__HIP_DEVICE_COMPILE__) && __has_builtin(__builtin_amdgcn_tensor_load_to_lds)
#define HAVE_TDM 1
#else
#define HAVE_TDM 0
#endif

#if defined(__HIP_DEVICE_COMPILE__) && __has_builtin(__builtin_amdgcn_s_wait_tensorcnt)
#define TDM_WAIT() __builtin_amdgcn_s_wait_tensorcnt(0)
#else
#define TDM_WAIT()
#endif

#if HAVE_TDM
typedef __attribute__((ext_vector_type(4))) unsigned int u32x4;
typedef __attribute__((ext_vector_type(4))) int i32x4;
typedef __attribute__((ext_vector_type(8))) int i32x8;

// Best-effort D# per CDNA5 ISA ch.8: 2D f16 tile (tile_w x tile_h) from a
// row-major tensor (tensor_w x tensor_h, stride_elems), landing at lds_off
// with LDS padding of 4 DWORDs every 16 DWORDs (matches LDP = GBK+8 halves).
__device__ inline void tdm_load_tile_f16(unsigned lds_off, const _Float16* gptr,
                                         unsigned tensor_w, unsigned tensor_h,
                                         unsigned tile_w, unsigned tile_h,
                                         unsigned stride_elems)
{
    unsigned long long ga = (unsigned long long)gptr;
    u32x4 g0;
    g0.x = 1u;                                    // count=1, user-mode D#
    g0.y = lds_off;                               // lds_addr (bytes)
    g0.z = (unsigned)ga;                          // global_addr[31:0]
    g0.w = (unsigned)((ga >> 32) & 0x01ffffffu) | 0x80000000u; // addr[56:32] | type=2
    i32x8 g1;
    g1[0] = (1 << 16)      // data_size = 2 bytes
          | (1 << 20)      // pad_enable
          | (3 << 22)      // pad_interval: 16 DWORDs (= 32 halves)
          | (3 << 25);     // pad_amount: 4 DWORDs (= 8 halves)
    g1[1] = (int)((tensor_w & 0xffffu) << 16);                      // dim0[15:0]
    g1[2] = (int)((tensor_w >> 16) | ((tensor_h & 0xffffu) << 16)); // dim0 hi | dim1 lo
    g1[3] = (int)((tensor_h >> 16) | (tile_w << 16));               // dim1 hi | tile_dim0
    g1[4] = (int)tile_h;                                            // tile_dim1 (tile_dim2=0)
    g1[5] = (int)stride_elems;                                      // dim0_stride lo
    g1[6] = 0;
    g1[7] = 0;
    i32x4 gz = {0, 0, 0, 0};
#if __clang_major__ >= 23
    i32x8 gz8 = {0, 0, 0, 0, 0, 0, 0, 0};
    __builtin_amdgcn_tensor_load_to_lds(g0, g1, gz, gz, gz8, 0);
#else
    __builtin_amdgcn_tensor_load_to_lds(g0, g1, gz, gz, 0);
#endif
}
#endif

// ------------------------------- generic WMMA GEMM --------------------------
// C[M,N] (f32, ldc) = A[M,K] (f16, lda) * B[K,N] (f16, ldb)
//                   + (Cin ? cinScale*Cin : 0) + (bias ? bias[col] : 0), act=1 => relu
// block = 128 threads = 4 waves; block tile 128x32; wave tile 32x32 (2x2 WMMA).
// Double-buffered K loop; requires K % 32 == 0 (true for all call sites).
__global__ void __launch_bounds__(128)
gemm_wmma_kernel(const _Float16* __restrict__ A, const _Float16* __restrict__ Bm,
                 float* __restrict__ C, int M, int N, int Kd,
                 int lda, int ldb, int ldc,
                 const float* __restrict__ Cin, float cinScale,
                 const float* __restrict__ bias, int act)
{
    __shared__ _Float16 As[2][GBM][LDP];
    __shared__ _Float16 Bs[2][GBN][LDP];
    const int tid  = threadIdx.x;
    const int lane = tid & 31;
    const int wv   = tid >> 5;
    const int hi   = lane >> 4;
    const int lr   = lane & 15;
    const int m0   = blockIdx.x * GBM;
    const int n0   = blockIdx.y * GBN;
    const bool fastB = (n0 + GBN <= N) && ((ldb & 7) == 0);
#if !HAVE_TDM
    const bool fastA = (m0 + GBM <= M) && ((lda & 7) == 0);
#endif

    v8f acc[2][2] = {};
    if (Cin) {
#pragma unroll
        for (int f = 0; f < 2; ++f)
#pragma unroll
            for (int r = 0; r < 8; ++r) {
                int row = m0 + wv * 32 + f * 16 + hi * 8 + r;
                int c0 = n0 + lr, c1 = n0 + 16 + lr;
                if (row < M) {
                    if (c0 < N) acc[f][0][r] = cinScale * Cin[(size_t)row * ldc + c0];
                    if (c1 < N) acc[f][1][r] = cinScale * Cin[(size_t)row * ldc + c1];
                }
            }
    }

    auto stageA = [&](int k0, int buf) {
#if HAVE_TDM
        if (tid == 0)
            tdm_load_tile_f16((unsigned)(unsigned long long)(void*)&As[buf][0][0],
                              A + (size_t)m0 * lda + k0,
                              (unsigned)Kd, (unsigned)M, GBK, GBM, (unsigned)lda);
#else
        if (fastA) {
            for (int e = tid; e < (GBM * GBK) / 8; e += 128) {
                int r = e >> 2, c8 = e & 3;
                *(v8h*)&As[buf][r][c8 * 8] =
                    *(const v8h*)&A[(size_t)(m0 + r) * lda + k0 + c8 * 8];
            }
        } else {
            for (int e = tid; e < GBM * GBK; e += 128) {
                int r = e >> 5, kk = e & 31;
                int gr = m0 + r, gk = k0 + kk;
                As[buf][r][kk] = (gr < M && gk < Kd) ? A[(size_t)gr * lda + gk] : (_Float16)0.f;
            }
        }
#endif
    };
    auto stageB = [&](int k0, int buf) {
        if (fastB) {
            for (int e = tid; e < (GBN * GBK) / 8; e += 128) {
                int k = e >> 2, c8 = e & 3;
                v8h bv = *(const v8h*)&Bm[(size_t)(k0 + k) * ldb + n0 + c8 * 8];
#pragma unroll
                for (int j = 0; j < 8; ++j) Bs[buf][c8 * 8 + j][k] = bv[j];
            }
        } else {
            for (int e = tid; e < GBN * GBK; e += 128) {
                int kk = e >> 5, c = e & 31;
                int gk = k0 + kk, gc = n0 + c;
                Bs[buf][c][kk] = (gk < Kd && gc < N) ? Bm[(size_t)gk * ldb + gc] : (_Float16)0.f;
            }
        }
    };

    stageA(0, 0); stageB(0, 0);
    TDM_WAIT();
    __syncthreads();
    int cur = 0;
    for (int k0 = 0; k0 < Kd; k0 += GBK) {
        int nxt = cur ^ 1;
        if (k0 + GBK < Kd) { stageA(k0 + GBK, nxt); stageB(k0 + GBK, nxt); }
        v16h af0, af1, bf0, bf1;
        {
            const _Float16* ap0 = &As[cur][wv * 32 + lr][0];
            const _Float16* ap1 = &As[cur][wv * 32 + 16 + lr][0];
#pragma unroll
            for (int j = 0; j < 8; ++j) {
                af0[j] = ap0[hi * 8 + j]; af0[8 + j] = ap0[16 + hi * 8 + j];
                af1[j] = ap1[hi * 8 + j]; af1[8 + j] = ap1[16 + hi * 8 + j];
            }
            const _Float16* bp0 = &Bs[cur][lr][0];
            const _Float16* bp1 = &Bs[cur][16 + lr][0];
#pragma unroll
            for (int j = 0; j < 8; ++j) {
                bf0[j] = bp0[hi * 8 + j]; bf0[8 + j] = bp0[16 + hi * 8 + j];
                bf1[j] = bp1[hi * 8 + j]; bf1[8 + j] = bp1[16 + hi * 8 + j];
            }
        }
#if defined(__HIP_DEVICE_COMPILE__)
        acc[0][0] = __builtin_amdgcn_wmma_f32_16x16x32_f16(false, af0, false, bf0, (short)0, acc[0][0], false, false);
        acc[0][1] = __builtin_amdgcn_wmma_f32_16x16x32_f16(false, af0, false, bf1, (short)0, acc[0][1], false, false);
        acc[1][0] = __builtin_amdgcn_wmma_f32_16x16x32_f16(false, af1, false, bf0, (short)0, acc[1][0], false, false);
        acc[1][1] = __builtin_amdgcn_wmma_f32_16x16x32_f16(false, af1, false, bf1, (short)0, acc[1][1], false, false);
#endif
        TDM_WAIT();
        __syncthreads();
        cur = nxt;
    }

#pragma unroll
    for (int f = 0; f < 2; ++f)
#pragma unroll
        for (int r = 0; r < 8; ++r) {
            int row = m0 + wv * 32 + f * 16 + hi * 8 + r;
            if (row >= M) continue;
            int c0 = n0 + lr, c1 = n0 + 16 + lr;
            float v0 = acc[f][0][r], v1 = acc[f][1][r];
            if (bias) { if (c0 < N) v0 += bias[c0]; if (c1 < N) v1 += bias[c1]; }
            if (act == 1) { v0 = fmaxf(v0, 0.f); v1 = fmaxf(v1, 0.f); }
            if (c0 < N) C[(size_t)row * ldc + c0] = v0;
            if (c1 < N) C[(size_t)row * ldc + c1] = v1;
        }
}

// ------------------- fused MAGAC: (sup[h,k] x xg[b]) . Wf -> out_h ----------
// grid: x = N/128, y = SEQ/32, z = b*16 + h*4 + k. All dims exact multiples.
__global__ void __launch_bounds__(128)
magac_fused_kernel(const _Float16* __restrict__ sup, const _Float16* __restrict__ xg,
                   const float* __restrict__ Wf, float* __restrict__ outh)
{
    __shared__ _Float16 As[2][GBM][LDP];
    __shared__ _Float16 Bs[2][GBN][LDP];
    const int tid  = threadIdx.x;
    const int lane = tid & 31;
    const int wv   = tid >> 5;
    const int hi   = lane >> 4;
    const int lr   = lane & 15;
    const int z    = blockIdx.z;
    const int b    = z >> 4;
    const int hk   = z & 15;
    const int h    = hk >> 2;
    const int kidx = hk & 3;
    const _Float16* A  = sup + (size_t)hk * (NN_ * NN_);
    const _Float16* Bm = xg + (size_t)b * (NN_ * SEQL);
    const int n0 = blockIdx.x * GBM;
    const int l0 = blockIdx.y * GBN;

    auto stageA = [&](int k0, int buf) {
#if HAVE_TDM
        if (tid == 0)
            tdm_load_tile_f16((unsigned)(unsigned long long)(void*)&As[buf][0][0],
                              A + (size_t)n0 * NN_ + k0,
                              NN_, NN_, GBK, GBM, NN_);
#else
        for (int e = tid; e < (GBM * GBK) / 8; e += 128) {
            int r = e >> 2, c8 = e & 3;
            *(v8h*)&As[buf][r][c8 * 8] = *(const v8h*)&A[(size_t)(n0 + r) * NN_ + k0 + c8 * 8];
        }
#endif
    };
    auto stageB = [&](int k0, int buf) {
        for (int e = tid; e < (GBN * GBK) / 8; e += 128) {
            int k = e >> 2, c8 = e & 3;
            v8h bv = *(const v8h*)&Bm[(size_t)(k0 + k) * SEQL + l0 + c8 * 8];
#pragma unroll
            for (int j = 0; j < 8; ++j) Bs[buf][c8 * 8 + j][k] = bv[j];
        }
    };

    v8f acc[2][2] = {};
    stageA(0, 0); stageB(0, 0);
    TDM_WAIT();
    __syncthreads();
    int cur = 0;
    for (int k0 = 0; k0 < NN_; k0 += GBK) {
        int nxt = cur ^ 1;
        if (k0 + GBK < NN_) { stageA(k0 + GBK, nxt); stageB(k0 + GBK, nxt); }
        v16h af0, af1, bf0, bf1;
        {
            const _Float16* ap0 = &As[cur][wv * 32 + lr][0];
            const _Float16* ap1 = &As[cur][wv * 32 + 16 + lr][0];
#pragma unroll
            for (int j = 0; j < 8; ++j) {
                af0[j] = ap0[hi * 8 + j]; af0[8 + j] = ap0[16 + hi * 8 + j];
                af1[j] = ap1[hi * 8 + j]; af1[8 + j] = ap1[16 + hi * 8 + j];
            }
            const _Float16* bp0 = &Bs[cur][lr][0];
            const _Float16* bp1 = &Bs[cur][16 + lr][0];
#pragma unroll
            for (int j = 0; j < 8; ++j) {
                bf0[j] = bp0[hi * 8 + j]; bf0[8 + j] = bp0[16 + hi * 8 + j];
                bf1[j] = bp1[hi * 8 + j]; bf1[8 + j] = bp1[16 + hi * 8 + j];
            }
        }
#if defined(__HIP_DEVICE_COMPILE__)
        acc[0][0] = __builtin_amdgcn_wmma_f32_16x16x32_f16(false, af0, false, bf0, (short)0, acc[0][0], false, false);
        acc[0][1] = __builtin_amdgcn_wmma_f32_16x16x32_f16(false, af0, false, bf1, (short)0, acc[0][1], false, false);
        acc[1][0] = __builtin_amdgcn_wmma_f32_16x16x32_f16(false, af1, false, bf0, (short)0, acc[1][0], false, false);
        acc[1][1] = __builtin_amdgcn_wmma_f32_16x16x32_f16(false, af1, false, bf1, (short)0, acc[1][1], false, false);
#endif
        TDM_WAIT();
        __syncthreads();
        cur = nxt;
    }

    // weighted reduce over l within tile, atomic accumulate into out_h[b,h,n]
#pragma unroll
    for (int f = 0; f < 2; ++f)
#pragma unroll
        for (int r = 0; r < 8; ++r) {
            int n = n0 + wv * 32 + f * 16 + hi * 8 + r;
            size_t wbase = (((size_t)h * NN_ + n) * (KCH + 1) + kidx) * SEQL;
            float p = acc[f][0][r] * Wf[wbase + l0 + lr] +
                      acc[f][1][r] * Wf[wbase + l0 + 16 + lr];
            p += __shfl_xor(p, 1, 32);
            p += __shfl_xor(p, 2, 32);
            p += __shfl_xor(p, 4, 32);
            p += __shfl_xor(p, 8, 32);
            if (lr == 0) atomicAdd(&outh[((size_t)(b * HH + h)) * NN_ + n], p);
        }
}

// ------------------------------ elementwise helpers -------------------------
__global__ void cvt_f16_kernel(const float* __restrict__ s, _Float16* __restrict__ d, int n)
{
    int i = blockIdx.x * 256 + threadIdx.x;
    if (i < n) d[i] = (_Float16)s[i];
}

// zero-padded conversion: d[rows x dcols] from s[rows x scols]
__global__ void cvt_pad_kernel(const float* __restrict__ s, _Float16* __restrict__ d,
                               int rows, int scols, int dcols)
{
    int i = blockIdx.x * 256 + threadIdx.x;
    if (i >= rows * dcols) return;
    int c = i % dcols, r = i / dcols;
    d[i] = (c < scols) ? (_Float16)s[(size_t)r * scols + c] : (_Float16)0.f;
}

__global__ void cvt_rev_kernel(const float* __restrict__ s, _Float16* __restrict__ d)
{
    int i = blockIdx.x * 256 + threadIdx.x;
    if (i >= BL_ * DM) return;
    int c = i & (DM - 1); int bl = i >> 9; int l = bl & (SEQL - 1); int b = bl >> 10;
    d[(((size_t)b * SEQL) + (SEQL - 1 - l)) * DM + c] = (_Float16)s[i];
}

__global__ void xg_cvt_kernel(const float* __restrict__ x, _Float16* __restrict__ xg)
{
    int i = blockIdx.x * 256 + threadIdx.x;
    if (i >= BB * DM * SEQL) return;
    int l = i & (SEQL - 1); int m = (i >> 10) & (DM - 1); int b = i >> 19;
    xg[i] = (_Float16)x[(((size_t)b * SEQL) + l) * DM + m];
}

__global__ void conv_silu_kernel(const float* __restrict__ xr,
                                 const float* __restrict__ cw, const float* __restrict__ cb,
                                 float* __restrict__ xp, _Float16* __restrict__ xp16)
{
    int i = blockIdx.x * 256 + threadIdx.x;
    if (i >= BL_ * DE_) return;
    int de = i & 63; int bl = i >> 6; int l = bl & (SEQL - 1); int b = bl >> 10;
    float acc = cb[de];
#pragma unroll
    for (int j = 0; j < 3; ++j) {
        int ls = l - 2 + j;
        if (ls >= 0) acc += xr[((size_t)(b * SEQL + ls)) * 128 + de] * cw[de * 3 + j];
    }
    float v = acc / (1.f + __expf(-acc));  // silu
    xp[i] = v; xp16[i] = (_Float16)v;
}

__global__ void delta_kernel(const float* __restrict__ xdbl,
                             const float* __restrict__ dtw, const float* __restrict__ dtb,
                             float* __restrict__ del)
{
    int i = blockIdx.x * 256 + threadIdx.x;
    if (i >= BL_ * DE_) return;
    int de = i & 63; int bl = i >> 6;
    const float* xd = xdbl + (size_t)bl * XDS;
    float a = dtb[de];
#pragma unroll
    for (int r = 0; r < DTR_; ++r) a += xd[r] * dtw[r * DE_ + de];
    del[i] = (a > 20.f) ? a : log1pf(__expf(a));  // softplus
}

// one wave per (b, de); lane owns dh=lane and dh=lane+32; shfl reduce over DH
__global__ void __launch_bounds__(32)
scan_kernel(const float* __restrict__ del, const float* __restrict__ xdbl,
            const float* __restrict__ xp, const float* __restrict__ xr,
            const float* __restrict__ Alog, const float* __restrict__ Dp,
            float* __restrict__ y, _Float16* __restrict__ y16)
{
    int bde = blockIdx.x;
    int b = bde >> 6, de = bde & 63;
    int lane = threadIdx.x;
    float A0 = -__expf(Alog[de * DH_ + lane]);
    float A1 = -__expf(Alog[de * DH_ + lane + 32]);
    float Dv = Dp[de];
    float s0 = 0.f, s1 = 0.f;
    for (int t = 0; t < SEQL; ++t) {
        size_t blt = (size_t)b * SEQL + t;
        float dl = del[blt * DE_ + de];
        float xpt = xp[blt * DE_ + de];
        const float* xd = xdbl + blt * XDS;
        float Bv0 = xd[DTR_ + lane],        Bv1 = xd[DTR_ + lane + 32];
        float Cv0 = xd[DTR_ + DH_ + lane],  Cv1 = xd[DTR_ + DH_ + lane + 32];
        s0 = __expf(dl * A0) * s0 + dl * Bv0 * xpt;
        s1 = __expf(dl * A1) * s1 + dl * Bv1 * xpt;
        float part = s0 * Cv0 + s1 * Cv1;
        for (int m = 16; m >= 1; m >>= 1) part += __shfl_xor(part, m, 32);
        if (lane == 0) {
            float res = xr[blt * 128 + 64 + de];
            float yv = (part + xpt * Dv) / (1.f + __expf(-res));  // * sigmoid(res)
            y[blt * DE_ + de] = yv;
            y16[blt * DE_ + de] = (_Float16)yv;
        }
    }
}

// out = LN(xin + ya + yb(flipped along seq)) ; ya/yb optional ; out16 optional
__global__ void __launch_bounds__(128)
add_ln_kernel(const float* __restrict__ xin, const float* __restrict__ ya,
              const float* __restrict__ yb,
              const float* __restrict__ g, const float* __restrict__ be,
              float* __restrict__ out, _Float16* __restrict__ out16)
{
    int bl = blockIdx.x; int b = bl >> 10; int l = bl & (SEQL - 1); int lf = SEQL - 1 - l;
    int tid = threadIdx.x;
    __shared__ float s1[128], s2[128];
    float v[4]; float sum = 0.f, sq = 0.f;
#pragma unroll
    for (int j = 0; j < 4; ++j) {
        int c = tid + j * 128;
        float t = xin[(size_t)bl * DM + c];
        if (ya) t += ya[(size_t)bl * DM + c];
        if (yb) t += yb[((size_t)b * SEQL + lf) * DM + c];
        v[j] = t; sum += t; sq += t * t;
    }
    s1[tid] = sum; s2[tid] = sq; __syncthreads();
    for (int s = 64; s; s >>= 1) {
        if (tid < s) { s1[tid] += s1[tid + s]; s2[tid] += s2[tid + s]; }
        __syncthreads();
    }
    float mu = s1[0] * (1.f / DM);
    float var = s2[0] * (1.f / DM) - mu * mu;
    float inv = rsqrtf(var + 1e-5f);
#pragma unroll
    for (int j = 0; j < 4; ++j) {
        int c = tid + j * 128;
        float o = (v[j] - mu) * inv * g[c] + be[c];
        out[(size_t)bl * DM + c] = o;
        if (out16) out16[(size_t)bl * DM + c] = (_Float16)o;
    }
}

// ------------------------------ MAGAC prep ----------------------------------
__global__ void magac_qk_kernel(const float* __restrict__ pe, const float* __restrict__ W,
                                float* __restrict__ Q)
{
    int i = blockIdx.x * 256 + threadIdx.x;
    if (i >= NN_ * HH * DEMB_) return;
    int dp = i % DEMB_; int h = (i / DEMB_) & 3; int n = i / (HH * DEMB_);
    float a = 0.f;
#pragma unroll
    for (int d = 0; d < DEMB_; ++d) a += pe[n * DEMB_ + d] * W[(d * HH + h) * DEMB_ + dp];
    Q[i] = a;  // [n][h][dp]
}

__global__ void __launch_bounds__(128)
magac_abase_kernel(const float* __restrict__ pe, const float* __restrict__ psi,
                   float* __restrict__ Abase)
{
    int n = blockIdx.x; int tid = threadIdx.x;
    __shared__ float sb[128];
    float pen[DEMB_];
#pragma unroll
    for (int d = 0; d < DEMB_; ++d) pen[d] = pe[n * DEMB_ + d];
    float psv = psi[0];
    float v[4]; float mx = -1e30f;
#pragma unroll
    for (int j = 0; j < 4; ++j) {
        int m = tid + j * 128;
        float d2 = 0.f;
#pragma unroll
        for (int d = 0; d < DEMB_; ++d) { float df = pen[d] - pe[m * DEMB_ + d]; d2 += df * df; }
        v[j] = __expf(-psv * d2);
        mx = fmaxf(mx, v[j]);
    }
    sb[tid] = mx; __syncthreads();
    for (int s = 64; s; s >>= 1) { if (tid < s) sb[tid] = fmaxf(sb[tid], sb[tid + s]); __syncthreads(); }
    mx = sb[0]; __syncthreads();
    float se = 0.f;
#pragma unroll
    for (int j = 0; j < 4; ++j) { v[j] = __expf(v[j] - mx); se += v[j]; }
    sb[tid] = se; __syncthreads();
    for (int s = 64; s; s >>= 1) { if (tid < s) sb[tid] += sb[tid + s]; __syncthreads(); }
    float inv = 1.f / sb[0];
#pragma unroll
    for (int j = 0; j < 4; ++j) Abase[(size_t)n * NN_ + tid + j * 128] = v[j] * inv;
}

__global__ void __launch_bounds__(128)
magac_attn_kernel(const float* __restrict__ Q, const float* __restrict__ Kq,
                  const float* __restrict__ Abase, const float* __restrict__ alpha_s,
                  float* __restrict__ Aeff, _Float16* __restrict__ sup,
                  _Float16* __restrict__ A2)
{
    int hn = blockIdx.x; int h = hn >> 9; int n = hn & (NN_ - 1); int tid = threadIdx.x;
    __shared__ float sb[128];
    float qn[DEMB_];
#pragma unroll
    for (int d = 0; d < DEMB_; ++d) qn[d] = Q[(n * HH + h) * DEMB_ + d];
    const float scale = 0.316227766016838f;  // 1/sqrt(DEMB)
    float v[4]; float mx = -1e30f;
#pragma unroll
    for (int j = 0; j < 4; ++j) {
        int m = tid + j * 128;
        float s = 0.f;
#pragma unroll
        for (int d = 0; d < DEMB_; ++d) s += qn[d] * Kq[(m * HH + h) * DEMB_ + d];
        v[j] = s * scale; mx = fmaxf(mx, v[j]);
    }
    sb[tid] = mx; __syncthreads();
    for (int s = 64; s; s >>= 1) { if (tid < s) sb[tid] = fmaxf(sb[tid], sb[tid + s]); __syncthreads(); }
    mx = sb[0]; __syncthreads();
    float se = 0.f;
#pragma unroll
    for (int j = 0; j < 4; ++j) { v[j] = __expf(v[j] - mx); se += v[j]; }
    sb[tid] = se; __syncthreads();
    for (int s = 64; s; s >>= 1) { if (tid < s) sb[tid] += sb[tid + s]; __syncthreads(); }
    float inv = 1.f / sb[0];
    float alpha = 1.f / (1.f + __expf(-alpha_s[0]));
#pragma unroll
    for (int j = 0; j < 4; ++j) {
        int m = tid + j * 128;
        float at = v[j] * inv;
        size_t o = (size_t)h * NN_ * NN_ + (size_t)n * NN_ + m;
        float ae = alpha * Abase[(size_t)n * NN_ + m] + (1.f - alpha) * at;
        Aeff[o] = ae;
        sup[((size_t)(h * (KCH + 1) + 1)) * NN_ * NN_ + (size_t)n * NN_ + m] = (_Float16)ae;
        A2[o] = (_Float16)(2.f * ae);
    }
}

__global__ void magac_eye_kernel(_Float16* __restrict__ sup, float* __restrict__ T0)
{
    int i = blockIdx.x * 256 + threadIdx.x;
    if (i >= HH * NN_ * NN_) return;
    int m = i & (NN_ - 1); int n = (i >> 9) & (NN_ - 1); int h = i >> 18;
    float v = (n == m) ? 1.f : 0.f;
    sup[((size_t)(h * (KCH + 1))) * NN_ * NN_ + (size_t)n * NN_ + m] = (_Float16)v;
    if (h == 0) T0[(size_t)n * NN_ + m] = v;
}

__global__ void wf_kernel(const float* __restrict__ pe, const float* __restrict__ Fw,
                          float* __restrict__ Wf)
{
    int i = blockIdx.x * 256 + threadIdx.x;
    if (i >= HH * NN_ * (KCH + 1) * SEQL) return;
    int l = i & (SEQL - 1); int k = (i >> 10) & 3; int n = (i >> 12) & (NN_ - 1); int h = i >> 21;
    float a = 0.f;
#pragma unroll
    for (int d = 0; d < DEMB_; ++d)
        a += pe[n * DEMB_ + d] * Fw[(((size_t)(h * DEMB_ + d)) * (KCH + 1) + k) * SEQL + l];
    Wf[i] = a;  // [h][n][k][l]
}

__global__ void outh_init_kernel(const float* __restrict__ pe, const float* __restrict__ fb,
                                 float* __restrict__ outh)
{
    int i = blockIdx.x * 256 + threadIdx.x;
    if (i >= BB * HH * NN_) return;
    int n = i & (NN_ - 1); int h = (i >> 9) & 3;
    float a = 0.f;
#pragma unroll
    for (int d = 0; d < DEMB_; ++d) a += pe[n * DEMB_ + d] * fb[h * DEMB_ + d];
    outh[i] = a;  // bf[h][n] broadcast over b
}

__global__ void __launch_bounds__(256)
final_kernel(const float* __restrict__ outh, const float* __restrict__ head_mix,
             const float* __restrict__ head_w, const float* __restrict__ head_b,
             float* __restrict__ out)
{
    int b = blockIdx.x; int tid = threadIdx.x;
    __shared__ float sred[256];
    float m0 = head_mix[0], m1 = head_mix[1], m2 = head_mix[2], m3 = head_mix[3];
    float mx = fmaxf(fmaxf(m0, m1), fmaxf(m2, m3));
    float e0 = __expf(m0 - mx), e1 = __expf(m1 - mx), e2 = __expf(m2 - mx), e3 = __expf(m3 - mx);
    float es = e0 + e1 + e2 + e3;
    float acc = 0.f, s2 = 0.f;
    for (int n = tid; n < NN_; n += 256) {
        float hw = head_w[n];
        float g = (outh[(size_t)(b * HH + 0) * NN_ + n] * e0 +
                   outh[(size_t)(b * HH + 1) * NN_ + n] * e1 +
                   outh[(size_t)(b * HH + 2) * NN_ + n] * e2 +
                   outh[(size_t)(b * HH + 3) * NN_ + n] * e3) / es;
        acc += g * hw;
        s2 += hw * hw;
    }
    sred[tid] = acc; __syncthreads();
    for (int s = 128; s; s >>= 1) { if (tid < s) sred[tid] += sred[tid + s]; __syncthreads(); }
    float mu = sred[0] + head_b[0];
    __syncthreads();
    sred[tid] = s2; __syncthreads();
    for (int s = 128; s; s >>= 1) { if (tid < s) sred[tid] += sred[tid + s]; __syncthreads(); }
    if (tid == 0) {
        out[b]      = mu;
        out[BB + b] = logf(1e-6f * sred[0] + 1e-6f);
    }
}

// ------------------------------- host driver --------------------------------
extern "C" void kernel_launch(void* const* d_in, const int* in_sizes, int n_in,
                              void* d_out, int out_size, void* d_ws, size_t ws_size,
                              hipStream_t stream)
{
    (void)in_sizes; (void)n_in; (void)out_size; (void)ws_size;
    const float* x_in     = (const float*)d_in[0];
    const float* m_in_w   = (const float*)d_in[1];
    const float* m_conv_w = (const float*)d_in[2];
    const float* m_conv_b = (const float*)d_in[3];
    const float* m_xproj_w= (const float*)d_in[4];
    const float* m_dt_w   = (const float*)d_in[5];
    const float* m_dt_b   = (const float*)d_in[6];
    const float* m_Alog   = (const float*)d_in[7];
    const float* m_D      = (const float*)d_in[8];
    const float* m_out_w  = (const float*)d_in[9];
    const float* ln1_g    = (const float*)d_in[10];
    const float* ln1_b    = (const float*)d_in[11];
    const float* ffn_w1   = (const float*)d_in[12];
    const float* ffn_b1   = (const float*)d_in[13];
    const float* ffn_w2   = (const float*)d_in[14];
    const float* ffn_b2   = (const float*)d_in[15];
    const float* ln2_g    = (const float*)d_in[16];
    const float* ln2_b    = (const float*)d_in[17];
    const float* psi_emb  = (const float*)d_in[18];
    const float* psi      = (const float*)d_in[19];
    const float* W_q      = (const float*)d_in[20];
    const float* W_k      = (const float*)d_in[21];
    const float* attn_a   = (const float*)d_in[22];
    const float* F_w      = (const float*)d_in[23];
    const float* f_b      = (const float*)d_in[24];
    const float* head_mix = (const float*)d_in[25];
    const float* head_w   = (const float*)d_in[26];
    const float* head_b   = (const float*)d_in[27];
    float* out = (float*)d_out;

    char* wp = (char*)d_ws;
    auto alloc = [&](size_t bytes) -> void* {
        void* p = (void*)wp;
        wp += (bytes + 255) & ~(size_t)255;
        return p;
    };
    const size_t NNB = (size_t)NN_ * NN_;

    float*    x_cur   = (float*)alloc((size_t)BL_ * DM * 4);
    _Float16* x16     = (_Float16*)alloc((size_t)BL_ * DM * 2);
    _Float16* xrev16  = (_Float16*)alloc((size_t)BL_ * DM * 2);
    _Float16* inw16   = (_Float16*)alloc((size_t)DM * 128 * 2);
    _Float16* xprojw16= (_Float16*)alloc((size_t)DE_ * XDS * 2);
    _Float16* outw16  = (_Float16*)alloc((size_t)DE_ * DM * 2);
    _Float16* fw1_16  = (_Float16*)alloc((size_t)DM * DU_ * 2);
    _Float16* fw2_16  = (_Float16*)alloc((size_t)DU_ * DM * 2);
    float*    xr[2];  float* xp[2]; _Float16* xp16[2]; float* xdbl[2];
    float*    dbuf[2]; float* ybuf[2]; _Float16* y16b[2]; float* yout[2];
    for (int d = 0; d < 2; ++d) {
        xr[d]   = (float*)alloc((size_t)BL_ * 128 * 4);
        xp[d]   = (float*)alloc((size_t)BL_ * DE_ * 4);
        xp16[d] = (_Float16*)alloc((size_t)BL_ * DE_ * 2);
        xdbl[d] = (float*)alloc((size_t)BL_ * XDS * 4);
        dbuf[d] = (float*)alloc((size_t)BL_ * DE_ * 4);
        ybuf[d] = (float*)alloc((size_t)BL_ * DE_ * 4);
        y16b[d] = (_Float16*)alloc((size_t)BL_ * DE_ * 2);
        yout[d] = (float*)alloc((size_t)BL_ * DM * 4);
    }
    float*    yln    = (float*)alloc((size_t)BL_ * DM * 4);
    _Float16* yln16  = (_Float16*)alloc((size_t)BL_ * DM * 2);
    float*    z1     = (float*)alloc((size_t)BL_ * DU_ * 4);
    _Float16* z1_16  = (_Float16*)alloc((size_t)BL_ * DU_ * 2);
    float*    zbuf   = (float*)alloc((size_t)BL_ * DM * 4);
    _Float16* xg16   = (_Float16*)alloc((size_t)BB * DM * SEQL * 2);
    float*    Qbuf   = (float*)alloc((size_t)NN_ * HH * DEMB_ * 4);
    float*    Kbuf   = (float*)alloc((size_t)NN_ * HH * DEMB_ * 4);
    float*    Abase  = (float*)alloc(NNB * 4);
    float*    Aeff   = (float*)alloc((size_t)HH * NNB * 4);
    _Float16* sup16  = (_Float16*)alloc((size_t)HH * (KCH + 1) * NNB * 2);
    _Float16* A2_16  = (_Float16*)alloc((size_t)HH * NNB * 2);
    float*    T0     = (float*)alloc(NNB * 4);
    float*    Tt     = (float*)alloc(NNB * 4);
    float*    Wf     = (float*)alloc((size_t)HH * NN_ * (KCH + 1) * SEQL * 4);
    float*    outh   = (float*)alloc((size_t)BB * HH * NN_ * 4);

    auto gemm = [&](const _Float16* A, const _Float16* Bm, float* C, int M, int N, int K,
                    int lda, int ldb, int ldc, const float* Cin, float cs,
                    const float* bias, int act) {
        dim3 g((M + GBM - 1) / GBM, (N + GBN - 1) / GBN, 1);
        gemm_wmma_kernel<<<g, 128, 0, stream>>>(A, Bm, C, M, N, K, lda, ldb, ldc, Cin, cs, bias, act);
    };
    auto cvt = [&](const float* s, _Float16* d, int n) {
        cvt_f16_kernel<<<(n + 255) / 256, 256, 0, stream>>>(s, d, n);
    };

    hipMemcpyAsync(x_cur, x_in, (size_t)BL_ * DM * 4, hipMemcpyDeviceToDevice, stream);

    for (int i = 0; i < 3; ++i) {
        cvt(x_cur, x16, BL_ * DM);
        cvt_rev_kernel<<<(BL_ * DM + 255) / 256, 256, 0, stream>>>(x_cur, xrev16);
        for (int d = 0; d < 2; ++d) {
            int sd = i * 2 + d;
            cvt(m_in_w + (size_t)sd * DM * 128, inw16, DM * 128);
            gemm(d ? xrev16 : x16, inw16, xr[d], BL_, 128, DM, DM, 128, 128, nullptr, 0.f, nullptr, 0);
            conv_silu_kernel<<<(BL_ * DE_ + 255) / 256, 256, 0, stream>>>(
                xr[d], m_conv_w + (size_t)sd * DE_ * 3, m_conv_b + (size_t)sd * DE_, xp[d], xp16[d]);
            cvt_pad_kernel<<<(DE_ * XDS + 255) / 256, 256, 0, stream>>>(
                m_xproj_w + (size_t)sd * DE_ * 132, xprojw16, DE_, 132, XDS);
            gemm(xp16[d], xprojw16, xdbl[d], BL_, XDS, DE_, DE_, XDS, XDS, nullptr, 0.f, nullptr, 0);
            delta_kernel<<<(BL_ * DE_ + 255) / 256, 256, 0, stream>>>(
                xdbl[d], m_dt_w + (size_t)sd * DTR_ * DE_, m_dt_b + (size_t)sd * DE_, dbuf[d]);
            scan_kernel<<<BB * DE_, 32, 0, stream>>>(
                dbuf[d], xdbl[d], xp[d], xr[d],
                m_Alog + (size_t)sd * DE_ * DH_, m_D + (size_t)sd * DE_, ybuf[d], y16b[d]);
            cvt(m_out_w + (size_t)sd * DE_ * DM, outw16, DE_ * DM);
            gemm(y16b[d], outw16, yout[d], BL_, DM, DE_, DE_, DM, DM, nullptr, 0.f, nullptr, 0);
        }
        add_ln_kernel<<<BL_, 128, 0, stream>>>(x_cur, yout[0], yout[1],
                                               ln1_g + i * DM, ln1_b + i * DM, yln, yln16);
        cvt(ffn_w1 + (size_t)i * DM * DU_, fw1_16, DM * DU_);
        gemm(yln16, fw1_16, z1, BL_, DU_, DM, DM, DU_, DU_, nullptr, 0.f, ffn_b1 + i * DU_, 1);
        cvt(z1, z1_16, BL_ * DU_);
        cvt(ffn_w2 + (size_t)i * DU_ * DM, fw2_16, DU_ * DM);
        gemm(z1_16, fw2_16, zbuf, BL_, DM, DU_, DU_, DM, DM, nullptr, 0.f, ffn_b2 + i * DM, 0);
        add_ln_kernel<<<BL_, 128, 0, stream>>>(yln, zbuf, nullptr,
                                               ln2_g + i * DM, ln2_b + i * DM, x_cur, nullptr);
    }

    // ---------------- MAGAC ----------------
    xg_cvt_kernel<<<(BB * DM * SEQL + 255) / 256, 256, 0, stream>>>(x_cur, xg16);
    magac_qk_kernel<<<(NN_ * HH * DEMB_ + 255) / 256, 256, 0, stream>>>(psi_emb, W_q, Qbuf);
    magac_qk_kernel<<<(NN_ * HH * DEMB_ + 255) / 256, 256, 0, stream>>>(psi_emb, W_k, Kbuf);
    magac_abase_kernel<<<NN_, 128, 0, stream>>>(psi_emb, psi, Abase);
    magac_eye_kernel<<<(HH * NN_ * NN_ + 255) / 256, 256, 0, stream>>>(sup16, T0);
    magac_attn_kernel<<<HH * NN_, 128, 0, stream>>>(Qbuf, Kbuf, Abase, attn_a, Aeff, sup16, A2_16);
    for (int h = 0; h < HH; ++h) {
        // T2 = 2A*T1 - I
        gemm(A2_16 + (size_t)h * NNB, sup16 + ((size_t)(h * 4 + 1)) * NNB, Tt,
             NN_, NN_, NN_, NN_, NN_, NN_, T0, -1.f, nullptr, 0);
        cvt(Tt, sup16 + ((size_t)(h * 4 + 2)) * NNB, (int)NNB);
        // T3 = 2A*T2 - T1
        gemm(A2_16 + (size_t)h * NNB, sup16 + ((size_t)(h * 4 + 2)) * NNB, Tt,
             NN_, NN_, NN_, NN_, NN_, NN_, Aeff + (size_t)h * NNB, -1.f, nullptr, 0);
        cvt(Tt, sup16 + ((size_t)(h * 4 + 3)) * NNB, (int)NNB);
    }
    wf_kernel<<<(HH * NN_ * (KCH + 1) * SEQL + 255) / 256, 256, 0, stream>>>(psi_emb, F_w, Wf);
    outh_init_kernel<<<(BB * HH * NN_ + 255) / 256, 256, 0, stream>>>(psi_emb, f_b, outh);
    magac_fused_kernel<<<dim3(NN_ / GBM, SEQL / GBN, BB * HH * (KCH + 1)), 128, 0, stream>>>(
        sup16, xg16, Wf, outh);
    final_kernel<<<BB, 256, 0, stream>>>(outh, head_mix, head_w, head_b, out);
}
